// ALIGNN_6201932775683
// MI455X (gfx1250) — matrix operand order repaired and checked by reading, not compiled, not run on hardware
//
#include <hip/hip_runtime.h>
#include <cstddef>
#include <cstdint>

typedef __bf16 bf16_t;
typedef __attribute__((ext_vector_type(16))) __bf16 v16bf;
typedef __attribute__((ext_vector_type(8)))  float  v8f;

#define Nn   1024
#define DEGc 12
#define Ec   (Nn*DEGc)      /* 12288  */
#define LGEc (Ec*DEGc)      /* 147456 */
#define Hc   256
#define EBc  80
#define TBc  40
#define EMBc 64
#define NLc  12

#define BM 128
#define BN 128
#define BK 32

// ---------- CDNA5 async global->LDS path (feature-detected, safe fallback) ----------
#if defined(__AMDGCN__) && __has_builtin(__builtin_amdgcn_global_load_async_to_lds_b128)
#define HAVE_ASYNC 1
typedef int v4i __attribute__((vector_size(16)));
typedef __attribute__((address_space(1))) v4i v4i_as1;
typedef __attribute__((address_space(3))) v4i v4i_as3;
__device__ __forceinline__ v4i_as1* to_gbl(const void* p){
  // flat address == global address on AMDGPU
  return (v4i_as1*)(unsigned long long)p;
}
__device__ __forceinline__ v4i_as3* to_lds(const void* p){
  // LDS flat addresses carry the LDS offset in the low 32 bits (aperture in high bits)
  return (v4i_as3*)(unsigned)(unsigned long long)p;
}
__device__ __forceinline__ void wait_async0(){
#if __has_builtin(__builtin_amdgcn_s_wait_asynccnt)
  __builtin_amdgcn_s_wait_asynccnt(0);
#else
  asm volatile("s_wait_asynccnt 0" ::: "memory");
#endif
}
#endif

__device__ __forceinline__ float sigmoidf_(float x){ return 1.f/(1.f+__expf(-x)); }

// ---------------- fp32 -> bf16 transpose conversion: out[n*K+k] = in[k*N+n], batched ----------------
__global__ void cvt_t_k(const float* __restrict__ in, bf16_t* __restrict__ out,
                        int K, int N, long total){
  long i = (long)blockIdx.x*blockDim.x + threadIdx.x;
  if (i >= total) return;
  long kn = (long)K*N;
  int s   = (int)(i / kn);
  int rem = (int)(i % kn);
  int k = rem / N, n = rem % N;
  out[(size_t)s*kn + (size_t)n*K + k] = (bf16_t)in[i];
}

// ---------------- atom embedding gather ----------------
__global__ void gather_k(const float* __restrict__ emb, const int* __restrict__ an,
                         float* __restrict__ x, int n){
  int i = blockIdx.x*blockDim.x + threadIdx.x;
  if (i < n) {
    int node = i / Hc, c = i % Hc;
    x[i] = emb[(size_t)an[node]*Hc + c];
  }
}

// ---------------- edge RBF (80 bins over [0,8]) + smooth cutoff ----------------
__global__ void edge_rbf_k(const float* __restrict__ r, float* __restrict__ rbf,
                           float* __restrict__ fcut, int n){ // n = Ec*EBc
  int i = blockIdx.x*blockDim.x + threadIdx.x;
  if (i >= n) return;
  int e = i / EBc, j = i % EBc;
  float x0 = r[3*e], x1 = r[3*e+1], x2 = r[3*e+2];
  float d = sqrtf(x0*x0 + x1*x1 + x2*x2);
  const float gamma = (EBc - 1) / 8.0f;
  float cj = 8.0f * j / (EBc - 1);
  float u = d - cj;
  rbf[i] = __expf(-gamma*u*u);
  if (j == 0) {
    float r2 = d*d;
    const float ron2 = 7.5f*7.5f, rc2 = 8.0f*8.0f;
    const float denom = (rc2 - ron2);
    float s = (rc2 - r2)*(rc2 - r2)*(rc2 + 2.f*r2 - 3.f*ron2)/(denom*denom*denom);
    fcut[e] = (d < 7.5f) ? 1.f : ((d < 8.0f) ? s : 0.f);
  }
}

// ---------------- angle RBF (40 bins over [-1,1]) ----------------
__global__ void angle_rbf_k(const float* __restrict__ r, const int* __restrict__ lgs,
                            const int* __restrict__ lgd, float* __restrict__ rbf, int n){
  int i = blockIdx.x*blockDim.x + threadIdx.x;
  if (i >= n) return;
  int t = i / TBc, j = i % TBc;
  int e1 = lgs[t], e2 = lgd[t];
  float a0 = -r[3*e1], a1 = -r[3*e1+1], a2 = -r[3*e1+2];
  float b0 =  r[3*e2], b1 =  r[3*e2+1], b2 =  r[3*e2+2];
  float dot = a0*b0 + a1*b1 + a2*b2;
  float na = sqrtf(a0*a0+a1*a1+a2*a2), nb = sqrtf(b0*b0+b1*b1+b2*b2);
  float c = fminf(1.f, fmaxf(-1.f, dot/(na*nb)));
  const float gamma = (TBc - 1) / 2.0f;
  float cj = -1.f + 2.f*j/(TBc-1);
  float u = c - cj;
  rbf[i] = __expf(-gamma*u*u);
}

// =====================================================================================
//  WMMA bf16 GEMM:  C[M,N] = A_f32[M,K] @ W[K,N] + bias,  W given transposed as Bt[N][K]
//  128x128 block, 8 waves (4x2), 32x64 per wave = 8 x v_wmma_f32_16x16x32_bf16 per K-step,
//  double-buffered LDS, async global->LDS weight stream when available.
// =====================================================================================
union U16  { uint4 u; bf16_t h[8]; };
union Frag { v16bf v; uint4 u[2]; };

template<bool ALIGNED>
__global__ __launch_bounds__(256) void gemm_bf16_k(
    const float* __restrict__ A, const bf16_t* __restrict__ Bt,
    const float* __restrict__ bias, float* __restrict__ C,
    int M, int N, int K)
{
  __shared__ __align__(16) bf16_t As[2][BM*BK];   // [row][k], 2 x 8 KB
  __shared__ __align__(16) bf16_t Bs[2][BN*BK];   // [col][k], 2 x 8 KB
  const int tid  = threadIdx.x;
  const int lane = tid & 31;
  const int wave = tid >> 5;
  const int wm   = wave >> 1;        // 0..3 -> 32-row strip
  const int wn   = wave & 1;         // 0..1 -> 64-col strip
  const int half = lane >> 4;
  const int l16  = lane & 15;
  const int mBase = blockIdx.x * BM;
  const int nBase = blockIdx.y * BN;

  // A staging: 128 rows x 32 k, 2 threads/row x 16 floats
  const int arow = tid >> 1;
  const int acol = (tid & 1) * 16;
  // B staging: 128 cols x 32 k, 2 threads/col x 16 bf16 (contiguous along K in Bt)
  const int bcol = tid >> 1;
  const int bkof = (tid & 1) * 16;

  v8f acc[2][4] = {};
  float aR[16];
  U16   bR[2];

  auto loadA = [&](int k0){
    if constexpr (ALIGNED) {
      const float4* ap = (const float4*)(A + (size_t)(mBase + arow)*K + k0 + acol);
      float4 v0 = ap[0], v1 = ap[1], v2 = ap[2], v3 = ap[3];
      aR[0]=v0.x; aR[1]=v0.y; aR[2]=v0.z; aR[3]=v0.w;
      aR[4]=v1.x; aR[5]=v1.y; aR[6]=v1.z; aR[7]=v1.w;
      aR[8]=v2.x; aR[9]=v2.y; aR[10]=v2.z; aR[11]=v2.w;
      aR[12]=v3.x; aR[13]=v3.y; aR[14]=v3.z; aR[15]=v3.w;
    } else {
      int gr = mBase + arow;
      #pragma unroll
      for (int j = 0; j < 16; ++j) {
        int kk = k0 + acol + j;
        aR[j] = (gr < M && kk < K) ? A[(size_t)gr*K + kk] : 0.f;
      }
    }
  };
  auto storeA = [&](int buf){
    bf16_t* p = &As[buf][arow*BK + acol];
    #pragma unroll
    for (int j = 0; j < 16; ++j) p[j] = (bf16_t)aR[j];
  };
  auto loadB = [&](int k0){           // sync path: global -> regs
    if constexpr (ALIGNED) {
      const bf16_t* g = Bt + (size_t)(nBase + bcol)*K + k0 + bkof;
      bR[0].u = *(const uint4*)(g);
      bR[1].u = *(const uint4*)(g + 8);
    } else {
      int col = nBase + bcol;
      #pragma unroll
      for (int i = 0; i < 2; ++i)
        #pragma unroll
        for (int j = 0; j < 8; ++j) {
          int kk = k0 + bkof + i*8 + j;
          bR[i].h[j] = (col < N && kk < K) ? Bt[(size_t)col*K + kk] : (bf16_t)0.f;
        }
    }
  };
  auto storeB = [&](int buf){
    bf16_t* p = &Bs[buf][bcol*BK + bkof];
    *(uint4*)(p)     = bR[0].u;
    *(uint4*)(p + 8) = bR[1].u;
  };
#ifdef HAVE_ASYNC
  auto stageB_async = [&](int k0, int buf){   // global -> LDS via ASYNCcnt DMA
    const bf16_t* g = Bt + (size_t)(nBase + bcol)*K + k0 + bkof;
    bf16_t* l = &Bs[buf][bcol*BK + bkof];
    __builtin_amdgcn_global_load_async_to_lds_b128(to_gbl(g),     to_lds(l),     0, 0);
    __builtin_amdgcn_global_load_async_to_lds_b128(to_gbl(g + 8), to_lds(l + 8), 0, 0);
  };
#endif

  const int nk = (K + BK - 1) / BK;

  // ---- prologue: stage tile 0 ----
  loadA(0); storeA(0);
#ifdef HAVE_ASYNC
  if constexpr (ALIGNED) { stageB_async(0, 0); wait_async0(); }
  else                   { loadB(0); storeB(0); }
#else
  loadB(0); storeB(0);
#endif
  __syncthreads();

  for (int t = 0; t < nk; ++t) {
    const int buf  = t & 1;
    const int nbuf = buf ^ 1;
    const bool more = (t + 1) < nk;
    const int k0n = (t + 1) * BK;

    if (more) {
      loadA(k0n);                       // global loads issued early
#ifdef HAVE_ASYNC
      if constexpr (ALIGNED) stageB_async(k0n, nbuf);   // DMA overlaps WMMA below
      else loadB(k0n);
#else
      loadB(k0n);
#endif
    }

    // ---- fragments (CDNA5 WMMA VGPR layouts) + 8 WMMA ----
    Frag af[2], bfr[4];
    #pragma unroll
    for (int mi = 0; mi < 2; ++mi) {
      int row = wm*32 + mi*16 + l16;
      const bf16_t* p = &As[buf][row*BK];
      af[mi].u[0] = *(const uint4*)(p + half*8);        // K = half*8 .. +7
      af[mi].u[1] = *(const uint4*)(p + 16 + half*8);   // K = 16+half*8 .. +7
    }
    #pragma unroll
    for (int ni = 0; ni < 4; ++ni) {
      int col = wn*64 + ni*16 + l16;
      const bf16_t* p = &Bs[buf][col*BK];
      bfr[ni].u[0] = *(const uint4*)(p + half*8);       // K = half*8 .. +7
      bfr[ni].u[1] = *(const uint4*)(p + 16 + half*8);  // K = 16+half*8 .. +7
    }
    #pragma unroll
    for (int mi = 0; mi < 2; ++mi)
      #pragma unroll
      for (int ni = 0; ni < 4; ++ni)
        acc[mi][ni] = __builtin_amdgcn_wmma_f32_16x16x32_bf16(
            false, af[mi].v, false, bfr[ni].v, (short)0, acc[mi][ni], false, false);

    if (more) {
      storeA(nbuf);
#ifdef HAVE_ASYNC
      if constexpr (ALIGNED) wait_async0();
      else storeB(nbuf);
#else
      storeB(nbuf);
#endif
    }
    __syncthreads();
  }

  // ---- epilogue: bias add + store (C layout: VGPR i -> M = half*8+i, N = l16) ----
  #pragma unroll
  for (int ni = 0; ni < 4; ++ni) {
    int colg = nBase + wn*64 + ni*16 + l16;
    if constexpr (!ALIGNED) { if (colg >= N) continue; }
    float bv = bias[colg];
    #pragma unroll
    for (int mi = 0; mi < 2; ++mi) {
      #pragma unroll
      for (int i = 0; i < 8; ++i) {
        int rowg = mBase + wm*32 + mi*16 + half*8 + i;
        if constexpr (ALIGNED) {
          C[(size_t)rowg*N + colg] = acc[mi][ni][i] + bv;
        } else {
          if (rowg < M) C[(size_t)rowg*N + colg] = acc[mi][ni][i] + bv;
        }
      }
    }
  }
}

// ---------------- fused [h-div] + LayerNorm + SiLU [+ residual], one row per block ----------------
__global__ void lnact_k(const float* __restrict__ in, const float* resid,
                        const float* num, const float* den,
                        const float* __restrict__ g, const float* __restrict__ b,
                        float* out, int W){
  __shared__ float sm[256];
  int r = blockIdx.x, c = threadIdx.x;
  size_t idx = (size_t)r*W + c;
  float t = in[idx];
  if (num) t += num[idx] / (den[idx] + 1e-6f);
  sm[c] = t; __syncthreads();
  for (int st = W>>1; st > 0; st >>= 1) { if (c < st) sm[c] += sm[c+st]; __syncthreads(); }
  float mu = sm[0] / W; __syncthreads();
  float dv = t - mu;
  sm[c] = dv*dv; __syncthreads();
  for (int st = W>>1; st > 0; st >>= 1) { if (c < st) sm[c] += sm[c+st]; __syncthreads(); }
  float var = sm[0] / W;
  float v = g[c] * dv * rsqrtf(var + 1e-5f) + b[c];
  float s = v * sigmoidf_(v);
  out[idx] = (resid ? resid[idx] : 0.f) + s;
}

// ---------------- edge gate + message + segment accumulation (atomics) ----------------
__global__ void edge_gate_k(const float* __restrict__ G0, const float* __restrict__ G1,
                            float* Mio, const float* __restrict__ Bh,
                            const int* __restrict__ s, const int* __restrict__ d,
                            const float* fc, float* num, float* den){
  int e = blockIdx.x, c = threadIdx.x;
  int se = s[e], de = d[e];
  size_t ei = (size_t)e*Hc + c;
  float m  = G0[(size_t)se*Hc + c] + G1[(size_t)de*Hc + c] + Mio[ei];
  float sg = sigmoidf_(m);
  if (fc) sg *= fc[e];
  float msg = sg * Bh[(size_t)se*Hc + c];
  atomicAdd(&num[(size_t)de*Hc + c], msg);
  atomicAdd(&den[(size_t)de*Hc + c], sg);
  Mio[ei] = m;
}

// ---------------- mean-pool over nodes + final linear ----------------
__global__ void final_k(const float* __restrict__ x, const float* __restrict__ fcW,
                        const float* __restrict__ fcb, float* __restrict__ out, int n){
  __shared__ float sm[Hc];
  int c = threadIdx.x;
  float s = 0.f;
  for (int i = 0; i < n; ++i) s += x[(size_t)i*Hc + c];
  sm[c] = (s / n) * fcW[c];
  __syncthreads();
  for (int st = Hc/2; st > 0; st >>= 1) { if (c < st) sm[c] += sm[c+st]; __syncthreads(); }
  if (c == 0) out[0] = sm[0] + fcb[0];
}

extern "C" void kernel_launch(void* const* d_in, const int* in_sizes, int n_in,
                              void* d_out, int out_size, void* d_ws, size_t ws_size,
                              hipStream_t stream)
{
  (void)in_sizes; (void)n_in; (void)out_size; (void)ws_size;
  const float* r        = (const float*)d_in[0];
  const float* atom_emb = (const float*)d_in[1];
  const float* eW1 = (const float*)d_in[2];
  const float* eb1 = (const float*)d_in[3];
  const float* eg1 = (const float*)d_in[4];
  const float* ebt1= (const float*)d_in[5];
  const float* eW2 = (const float*)d_in[6];
  const float* eb2 = (const float*)d_in[7];
  const float* eg2 = (const float*)d_in[8];
  const float* ebt2= (const float*)d_in[9];
  const float* aW1 = (const float*)d_in[10];
  const float* ab1 = (const float*)d_in[11];
  const float* ag1 = (const float*)d_in[12];
  const float* abt1= (const float*)d_in[13];
  const float* aW2 = (const float*)d_in[14];
  const float* ab2 = (const float*)d_in[15];
  const float* ag2 = (const float*)d_in[16];
  const float* abt2= (const float*)d_in[17];
  const float* gate_W = (const float*)d_in[18];
  const float* gate_b = (const float*)d_in[19];
  const float* upd_W  = (const float*)d_in[20];
  const float* upd_b  = (const float*)d_in[21];
  const float* ln_g   = (const float*)d_in[22];
  const float* ln_b   = (const float*)d_in[23];
  const float* fc_W   = (const float*)d_in[24];
  const float* fc_b   = (const float*)d_in[25];
  const int* an   = (const int*)d_in[26];
  const int* src  = (const int*)d_in[27];
  const int* dst  = (const int*)d_in[28];
  const int* lgs  = (const int*)d_in[29];
  const int* lgd  = (const int*)d_in[30];

  // ---- workspace carve-out ----
  char* base = (char*)d_ws;
  size_t off = 0;
  auto alloc = [&](size_t bytes) -> char* {
    char* p = base + off;
    off += (bytes + 255) & ~(size_t)255;
    return p;
  };
  bf16_t* wg_bf = (bf16_t*)alloc((size_t)NLc*3*Hc*Hc*2);   // transposed [N][K] per slice
  bf16_t* wu_bf = (bf16_t*)alloc((size_t)NLc*2*Hc*Hc*2);
  bf16_t* we1   = (bf16_t*)alloc((size_t)EBc*EMBc*2);
  bf16_t* we2   = (bf16_t*)alloc((size_t)EMBc*Hc*2);
  bf16_t* wa1   = (bf16_t*)alloc((size_t)TBc*EMBc*2);
  bf16_t* wa2   = (bf16_t*)alloc((size_t)EMBc*Hc*2);
  float* x   = (float*)alloc((size_t)Nn*Hc*4);
  float* y   = (float*)alloc((size_t)Ec*Hc*4);
  float* z   = (float*)alloc((size_t)LGEc*Hc*4);
  float* G0  = (float*)alloc((size_t)Ec*Hc*4);
  float* G1  = (float*)alloc((size_t)Ec*Hc*4);
  float* Bh  = (float*)alloc((size_t)Ec*Hc*4);
  float* A0  = (float*)alloc((size_t)Ec*Hc*4);
  float* num = (float*)alloc((size_t)Ec*Hc*4);
  float* den = (float*)alloc((size_t)Ec*Hc*4);
  float* big = (float*)alloc((size_t)LGEc*Hc*4);   // m-buffer; aliased rbf_a + t_a during embedding
  float* fcut  = (float*)alloc((size_t)Ec*4);
  float* rbf_e = (float*)alloc((size_t)Ec*EBc*4);
  float* t_e   = (float*)alloc((size_t)Ec*EMBc*4);
  float* rbf_a = big;                              // LGE*40 floats
  float* t_a   = big + (size_t)LGEc*TBc;           // LGE*64 floats (still inside big)

  auto gemm = [&](const float* Ap, const bf16_t* Bt, const float* biasp, float* Cp,
                  int M, int N, int K){
    dim3 g((M+BM-1)/BM, (N+BN-1)/BN);
    bool aligned = (M % BM == 0) && (N % BN == 0) && (K % BK == 0);
    if (aligned) gemm_bf16_k<true ><<<g, 256, 0, stream>>>(Ap, Bt, biasp, Cp, M, N, K);
    else         gemm_bf16_k<false><<<g, 256, 0, stream>>>(Ap, Bt, biasp, Cp, M, N, K);
  };
  auto lnact = [&](const float* in, const float* resid, const float* nump, const float* denp,
                   const float* g, const float* b, float* out, int rows, int W){
    lnact_k<<<rows, W, 0, stream>>>(in, resid, nump, denp, g, b, out, W);
  };
  auto cvtT = [&](const float* in, bf16_t* out, int K, int N, long total){
    cvt_t_k<<<(int)((total + 255) / 256), 256, 0, stream>>>(in, out, K, N, total);
  };

  // ---- weights -> bf16, transposed to [N][K] (once per launch) ----
  cvtT(gate_W, wg_bf, Hc, Hc, (long)NLc*3*Hc*Hc);
  cvtT(upd_W,  wu_bf, Hc, Hc, (long)NLc*2*Hc*Hc);
  cvtT(eW1, we1, EBc, EMBc, (long)EBc*EMBc);
  cvtT(eW2, we2, EMBc, Hc,  (long)EMBc*Hc);
  cvtT(aW1, wa1, TBc, EMBc, (long)TBc*EMBc);
  cvtT(aW2, wa2, EMBc, Hc,  (long)EMBc*Hc);

  // ---- embeddings ----
  gather_k<<<(Nn*Hc+255)/256, 256, 0, stream>>>(atom_emb, an, x, Nn*Hc);
  edge_rbf_k<<<(Ec*EBc+255)/256, 256, 0, stream>>>(r, rbf_e, fcut, Ec*EBc);
  gemm(rbf_e, we1, eb1, t_e, Ec, EMBc, EBc);
  lnact(t_e, nullptr, nullptr, nullptr, eg1, ebt1, t_e, Ec, EMBc);
  gemm(t_e, we2, eb2, y, Ec, Hc, EMBc);
  lnact(y, nullptr, nullptr, nullptr, eg2, ebt2, y, Ec, Hc);

  angle_rbf_k<<<(LGEc*TBc+255)/256, 256, 0, stream>>>(r, lgs, lgd, rbf_a, LGEc*TBc);
  gemm(rbf_a, wa1, ab1, t_a, LGEc, EMBc, TBc);
  lnact(t_a, nullptr, nullptr, nullptr, ag1, abt1, t_a, LGEc, EMBc);
  gemm(t_a, wa2, ab2, z, LGEc, Hc, EMBc);
  lnact(z, nullptr, nullptr, nullptr, ag2, abt2, z, LGEc, Hc);

  // ---- EGGC layer ----
  auto eggc = [&](int li, float* xn, int M1, float* ye, int M2,
                  const int* s, const int* d, const float* fc){
    const bf16_t* gW = wg_bf + (size_t)li*3*Hc*Hc;
    const bf16_t* uW = wu_bf + (size_t)li*2*Hc*Hc;
    const float*  gb = gate_b + (size_t)li*3*Hc;
    const float*  ub = upd_b  + (size_t)li*2*Hc;
    gemm(xn, gW,                    gb,        G0, M1, Hc, Hc);
    gemm(xn, gW + (size_t)Hc*Hc,    gb + Hc,   G1, M1, Hc, Hc);
    gemm(xn, uW + (size_t)Hc*Hc,    ub + Hc,   Bh, M1, Hc, Hc);
    gemm(xn, uW,                    ub,        A0, M1, Hc, Hc);
    gemm(ye, gW + (size_t)2*Hc*Hc,  gb + 2*Hc, big, M2, Hc, Hc);
    (void)hipMemsetAsync(num, 0, (size_t)M1*Hc*4, stream);
    (void)hipMemsetAsync(den, 0, (size_t)M1*Hc*4, stream);
    edge_gate_k<<<M2, Hc, 0, stream>>>(G0, G1, big, Bh, s, d, fc, num, den);
    lnact(A0, xn, num, den,
          ln_g + (size_t)(li*2)*Hc,   ln_b + (size_t)(li*2)*Hc,   xn, M1, Hc);
    lnact(big, ye, nullptr, nullptr,
          ln_g + (size_t)(li*2+1)*Hc, ln_b + (size_t)(li*2+1)*Hc, ye, M2, Hc);
  };

  for (int l = 0; l < 4; ++l) {          // ALIGNN: g-conv then lg-conv
    eggc(2*l,   x, Nn, y, Ec,   src, dst, fcut);
    eggc(2*l+1, y, Ec, z, LGEc, lgs, lgd, nullptr);
  }
  for (int l = 0; l < 4; ++l)            // GCN layers
    eggc(8+l, x, Nn, y, Ec, src, dst, fcut);

  final_k<<<1, Hc, 0, stream>>>(x, fc_W, fc_b, (float*)d_out, Nn);
}
